// NekoSpatialTransformDeformConv3x3_53601191854422
// MI455X (gfx1250) — compile-verified, matrix operand order checked by reading, not run on hardware
//
#include <hip/hip_runtime.h>
#include <hip/hip_fp16.h>

typedef _Float16 v8h  __attribute__((ext_vector_type(8)));
typedef _Float16 v16h __attribute__((ext_vector_type(16)));
typedef float    v8f  __attribute__((ext_vector_type(8)));

#define N_IMG 8
#define C_IN  128
#define H_DIM 64
#define W_DIM 64
#define O_OUT 128
#define HW    (H_DIM * W_DIM)            // 4096
#define CK    (C_IN * 9)                 // 1152
#define NPIX  (N_IMG * HW)               // 32768
#define LDS_A_STRIDE 296                 // 288 K-chunk + pad, keeps 16B alignment
#define P_STRIDE 72                      // dense: enables straight async memcpy

// ---------------------------------------------------------------------------
// Kernel 1: per-pixel spatial-transform -> 9 bilinear taps
// params[(pid*9+k)*8 + {0..3}] = clamped gather offsets (int, within one HxW image)
// params[(pid*9+k)*8 + {4..7}] = validity-folded bilinear weights (float bits)
// ---------------------------------------------------------------------------
__global__ __launch_bounds__(256) void offsets_kernel(
    const float* __restrict__ in,        // (N,C,H,W)
    const float* __restrict__ ste_w,     // (3,C)
    const float* __restrict__ ste_b,     // (3)
    const float* __restrict__ bn_gamma, const float* __restrict__ bn_beta,
    const float* __restrict__ bn_mean,  const float* __restrict__ bn_var,
    int* __restrict__ params)
{
    int pid = blockIdx.x * 256 + threadIdx.x;          // 0..NPIX-1
    int n  = pid >> 12;
    int hw = pid & 4095;
    int h  = hw >> 6;
    int w  = hw & 63;

    // 1x1 conv C -> 3 (coalesced across hw)
    const float* base = in + (size_t)n * C_IN * HW + hw;
    float s0 = 0.f, s1 = 0.f, s2 = 0.f;
    for (int c = 0; c < C_IN; ++c) {
        float x = base[(size_t)c * HW];
        s0 += x * ste_w[c];
        s1 += x * ste_w[C_IN + c];
        s2 += x * ste_w[2 * C_IN + c];
    }
    // bias + BN (folded)
    const float eps = 1e-5f;
    float sc0 = bn_gamma[0] * rsqrtf(bn_var[0] + eps);
    float sc1 = bn_gamma[1] * rsqrtf(bn_var[1] + eps);
    float sc2 = bn_gamma[2] * rsqrtf(bn_var[2] + eps);
    s0 = (s0 + ste_b[0] - bn_mean[0]) * sc0 + bn_beta[0];
    s1 = (s1 + ste_b[1] - bn_mean[1]) * sc1 + bn_beta[1];
    s2 = (s2 + ste_b[2] - bn_mean[2]) * sc2 + bn_beta[2];

    const float T_HARD  = (float)(3.14 / 6.0);
    const float S_RANGE = 0.75f;
    float theta = tanhf(s0) * T_HARD;
    float g1 = tanhf(s1) * S_RANGE + S_RANGE;
    float g2 = tanhf(s2) * S_RANGE + S_RANGE;
    float sn = sinf(theta), cs = cosf(theta);
    float a00 = cs * g1;
    float a11 = cs * g2;

    // OFF table exactly as in the reference (note row 2 repeats (-1,-1))
    const float OFF0[9] = {-1.f,-1.f,-1.f, 0.f,0.f,0.f, 1.f,1.f,1.f};
    const float OFF1[9] = {-1.f, 0.f,-1.f,-1.f,0.f,1.f,-1.f,0.f,1.f};

    int* pbase = params + (size_t)pid * 72;
    #pragma unroll
    for (int k = 0; k < 9; ++k) {
        float off_y =  OFF0[k] * a00 + OFF1[k] * sn;
        float off_x = -OFF0[k] * sn  + OFF1[k] * a11;
        float fy = (float)h + (float)(k / 3 - 1) + off_y;
        float fx = (float)w + (float)(k % 3 - 1) + off_x;
        float y0f = floorf(fy), x0f = floorf(fx);
        float wy = fy - y0f, wx = fx - x0f;
        int y0 = (int)y0f, x0 = (int)x0f;
        int y1 = y0 + 1,   x1 = x0 + 1;
        bool vy0 = (y0 >= 0) & (y0 < H_DIM);
        bool vy1 = (y1 >= 0) & (y1 < H_DIM);
        bool vx0 = (x0 >= 0) & (x0 < W_DIM);
        bool vx1 = (x1 >= 0) & (x1 < W_DIM);
        int cy0 = min(max(y0, 0), H_DIM - 1);
        int cy1 = min(max(y1, 0), H_DIM - 1);
        int cx0 = min(max(x0, 0), W_DIM - 1);
        int cx1 = min(max(x1, 0), W_DIM - 1);

        int4 addrs;
        addrs.x = cy0 * W_DIM + cx0;
        addrs.y = cy0 * W_DIM + cx1;
        addrs.z = cy1 * W_DIM + cx0;
        addrs.w = cy1 * W_DIM + cx1;
        float4 wts;
        wts.x = (1.f - wy) * (1.f - wx) * (float)(vy0 & vx0);
        wts.y = (1.f - wy) * wx         * (float)(vy0 & vx1);
        wts.z = wy * (1.f - wx)         * (float)(vy1 & vx0);
        wts.w = wy * wx                 * (float)(vy1 & vx1);

        *(int4*)  (pbase + k * 8)     = addrs;
        *(float4*)(pbase + k * 8 + 4) = wts;
    }
}

// ---------------------------------------------------------------------------
// Kernel 2: dconv_w (O, C*9) f32 -> f16 (already the GEMM-B layout: row o, col c*9+k)
// ---------------------------------------------------------------------------
__global__ __launch_bounds__(256) void wcvt_kernel(
    const float* __restrict__ w, _Float16* __restrict__ wh, int n)
{
    int i = blockIdx.x * 256 + threadIdx.x;
    if (i < n) wh[i] = (_Float16)w[i];
}

// ---------------------------------------------------------------------------
// Kernel 3: deformable-conv GEMM via v_wmma_f32_16x16x32_f16.
// Block = 256 threads (8 waves) = 16 pixels (fixed n,h, 16 consecutive w) x 128 outputs.
// Wave 'wv' computes the 16x16 tile (pixels x outputs[wv*16 .. wv*16+15]).
// K = 1152, processed in 4 c-chunks of 32 channels (K-chunk = 288 = 9 * 32).
// Param tile is staged with GLOBAL_LOAD_ASYNC_TO_LDS_B128 (ASYNCcnt path).
// Bilinear gathers use a uniform SGPR base + 32-bit unsigned element offsets
// so LLVM selects the GVS (saddr + voffset) addressing mode.
// ---------------------------------------------------------------------------
__global__ __launch_bounds__(256) void deform_gemm_kernel(
    const float*    __restrict__ in,     // (N,C,H,W)
    const int*      __restrict__ params, // per-pixel taps from kernel 1
    const _Float16* __restrict__ wt,     // (O, CK) f16
    const float*    __restrict__ bias,   // (O)
    float*          __restrict__ out)    // (N,O,H,W)
{
    __shared__ __align__(16) _Float16 a_lds[16 * LDS_A_STRIDE];
    __shared__ __align__(16) int p_lds[16 * P_STRIDE];

    const int t    = threadIdx.x;
    const int lane = t & 31;
    const int wv   = t >> 5;

    const int bid = blockIdx.x;              // N*H*(W/16) = 2048 blocks
    const int n   = bid >> 8;
    const int rem = bid & 255;
    const int h   = rem >> 2;
    const int w0  = (rem & 3) << 4;
    const int base_pid = (n << 12) + (h << 6) + w0;

    // ---- async-stage per-pixel sampling params: contiguous 4608B global -> LDS
    {
        const char* src = (const char*)(params + (size_t)base_pid * 72);
        for (int idx = t; idx < (16 * 72 * 4) / 16; idx += 256) {   // 288 x 16B
            unsigned lds_off = (unsigned)(uintptr_t)p_lds + (unsigned)(idx * 16);
            const void* gaddr = src + idx * 16;
            asm volatile("global_load_async_to_lds_b128 %0, %1, off"
                         :: "v"(lds_off), "v"(gaddr) : "memory");
        }
        asm volatile("s_wait_asynccnt 0" ::: "memory");
    }
    __syncthreads();

    // staging roles: lanes 0-15 of each half-wave cover consecutive pixels, same c
    const int p_me  = t & 15;       // pixel
    const int ci_me = t >> 4;       // 0..15 channel slot within chunk
    const float* __restrict__ inp_n = in + (size_t)n * C_IN * HW;  // uniform base

    // compute roles (WMMA f16 A/B lane layout)
    const int prow = lane & 15;                 // A row (pixel) / B col (output)
    const int koff = (lane >> 4) << 3;          // K sub-offset: 0 or 8
    const _Float16* abase = &a_lds[prow * LDS_A_STRIDE + koff];

    v8f acc = {0.f, 0.f, 0.f, 0.f, 0.f, 0.f, 0.f, 0.f};

    for (int c0 = 0; c0 < C_IN; c0 += 32) {
        // ---- build bilinear-sampled A tile: 16 pixels x (32 c * 9 k) halves
        const int* pp = &p_lds[p_me * P_STRIDE];
        #pragma unroll
        for (int cc = 0; cc < 2; ++cc) {
            int crel = ci_me + cc * 16;                     // 0..31
            unsigned cbase = (unsigned)(c0 + crel) * (unsigned)HW; // elem offset
            _Float16* dst = &a_lds[p_me * LDS_A_STRIDE + crel * 9];
            #pragma unroll
            for (int k = 0; k < 9; ++k) {
                unsigned i00 = cbase + (unsigned)pp[k * 8 + 0];
                unsigned i01 = cbase + (unsigned)pp[k * 8 + 1];
                unsigned i10 = cbase + (unsigned)pp[k * 8 + 2];
                unsigned i11 = cbase + (unsigned)pp[k * 8 + 3];
                float w00 = __int_as_float(pp[k * 8 + 4]);
                float w01 = __int_as_float(pp[k * 8 + 5]);
                float w10 = __int_as_float(pp[k * 8 + 6]);
                float w11 = __int_as_float(pp[k * 8 + 7]);
                float v = w00 * inp_n[i00] + w01 * inp_n[i01]
                        + w10 * inp_n[i10] + w11 * inp_n[i11];
                dst[k] = (_Float16)v;
            }
        }
        __syncthreads();

        // ---- 9 WMMA steps over this 288-wide K chunk
        const _Float16* wrow =
            wt + (size_t)(wv * 16 + prow) * CK + c0 * 9 + koff;
        if (c0 + 32 < C_IN)
            __builtin_prefetch(wrow + 288, 0, 1);   // next chunk's B fragment

        #pragma unroll
        for (int kk = 0; kk < 9; ++kk) {
            int kb = kk * 32;
            v8h alo = *(const v8h*)(abase + kb);        // K koff+0..7
            v8h ahi = *(const v8h*)(abase + kb + 16);   // K koff+16..23
            v8h blo = *(const v8h*)(wrow  + kb);
            v8h bhi = *(const v8h*)(wrow  + kb + 16);
            v16h a = __builtin_shufflevector(alo, ahi,
                0,1,2,3,4,5,6,7,8,9,10,11,12,13,14,15);
            v16h b = __builtin_shufflevector(blo, bhi,
                0,1,2,3,4,5,6,7,8,9,10,11,12,13,14,15);
            acc = __builtin_amdgcn_wmma_f32_16x16x32_f16(
                false, a, false, b, (short)0, acc, false, false);
        }
        __syncthreads();
    }

    // ---- epilogue: C/D layout -> (n, o, h, w0 + m); add bias
    const int o = wv * 16 + prow;
    const float bo = bias[o];
    float* obase = out + ((size_t)(n * O_OUT + o) * H_DIM + h) * W_DIM
                       + w0 + ((lane >> 4) << 3);     // lanes>=16 hold M=8..15
    float4 lo, hi;
    lo.x = acc[0] + bo; lo.y = acc[1] + bo; lo.z = acc[2] + bo; lo.w = acc[3] + bo;
    hi.x = acc[4] + bo; hi.y = acc[5] + bo; hi.z = acc[6] + bo; hi.w = acc[7] + bo;
    *(float4*)(obase)     = lo;
    *(float4*)(obase + 4) = hi;
}

// ---------------------------------------------------------------------------
extern "C" void kernel_launch(void* const* d_in, const int* in_sizes, int n_in,
                              void* d_out, int out_size, void* d_ws, size_t ws_size,
                              hipStream_t stream) {
    const float* input_feat = (const float*)d_in[0];
    const float* ste_w      = (const float*)d_in[1];
    const float* ste_b      = (const float*)d_in[2];
    const float* bn_gamma   = (const float*)d_in[3];
    const float* bn_beta    = (const float*)d_in[4];
    const float* bn_mean    = (const float*)d_in[5];
    const float* bn_var     = (const float*)d_in[6];
    const float* dconv_w    = (const float*)d_in[7];
    const float* dconv_b    = (const float*)d_in[8];
    float* out = (float*)d_out;

    // workspace layout: [params: NPIX*72 dwords][f16 weights: O*CK halves]
    int*       params = (int*)d_ws;
    _Float16*  wt_h   = (_Float16*)((char*)d_ws + (size_t)NPIX * 72 * sizeof(int));

    offsets_kernel<<<NPIX / 256, 256, 0, stream>>>(
        input_feat, ste_w, ste_b, bn_gamma, bn_beta, bn_mean, bn_var, params);

    const int nw = O_OUT * CK;   // 147456
    wcvt_kernel<<<(nw + 255) / 256, 256, 0, stream>>>(dconv_w, wt_h, nw);

    deform_gemm_kernel<<<N_IMG * H_DIM * (W_DIM / 16), 256, 0, stream>>>(
        input_feat, params, wt_h, dconv_b, out);
}
